// LSTMEncoder_41068477284726
// MI455X (gfx1250) — matrix operand order repaired and checked by reading, not compiled, not run on hardware
//
#include <hip/hip_runtime.h>
#include <hip/hip_bf16.h>
#include <math.h>

typedef __attribute__((ext_vector_type(16))) _Float16 v16h;
typedef __attribute__((ext_vector_type(8)))  float    v8f;

#define S_LEN   64
#define B_SZ    64
#define W_LEN   16
#define DE      300
#define DC      64
#define HC      128
#define HW      512
#define DIN     556
#define DIN_PAD 576      // pad K to multiple of 32 for 16x16x32 WMMA
#define NCH     4096     // S*B
#define GW      (4*HW)   // 2048 word-LSTM gate width
#define GC      (4*HC)   // 512  char-LSTM gate width

__device__ __forceinline__ float sigf(float x) { return 1.0f / (1.0f + __expf(-x)); }

// ---- WMMA 16x16x32 f16 tile loaders (ISA 7.12.2 layouts) ----
// A (16x32 MxK) from row-major source, pitch in halves.
// lane l: row m = l&15 ; K halves = kb..kb+7 (elems 0..7), kb+16..kb+23 (elems 8..15), kb = (l>>4)*8
__device__ __forceinline__ v16h load_a(const _Float16* tile, int pitch, int lane) {
  const int m  = lane & 15;
  const int kb = (lane >> 4) << 3;
  const _Float16* p = tile + m * pitch + kb;
  v16h r;
  *(float4*)&r       = *(const float4*)(p);
  *((float4*)&r + 1) = *(const float4*)(p + 16);
  return r;
}
// B (32x16 KxN) loaded from B^T (row-major weight matrix: row = output col, contiguous K).
// lane l: col n = l&15 ; K halves = kb..kb+7 (elems 0..7), kb+16..kb+23 (elems 8..15), kb = (l>>4)*8
__device__ __forceinline__ v16h load_b(const _Float16* bt, int pitch, int lane) {
  const int n  = lane & 15;
  const int kb = (lane >> 4) << 3;
  const _Float16* p = bt + n * pitch + kb;
  v16h r;
  *(float4*)&r       = *(const float4*)(p);
  *((float4*)&r + 1) = *(const float4*)(p + 16);
  return r;
}

// ---------------- utility kernels ----------------
__global__ void k_cvt(const float* __restrict__ src, _Float16* __restrict__ dst, int n) {
  int i = blockIdx.x * blockDim.x + threadIdx.x;
  const int stride = gridDim.x * blockDim.x;
  for (; i < n; i += stride) dst[i] = (_Float16)src[i];
}

// wWih (2048,556) f32 -> (2048,576) f16, zero-padded K
__global__ void k_pack_wWih(const float* __restrict__ w, _Float16* __restrict__ dst) {
  int i = blockIdx.x * blockDim.x + threadIdx.x;
  const int stride = gridDim.x * blockDim.x;
  const int total = GW * DIN_PAD;
  for (; i < total; i += stride) {
    const int r = i / DIN_PAD, c = i - r * DIN_PAD;
    dst[i] = (c < DIN) ? (_Float16)w[r * DIN + c] : (_Float16)0.0f;
  }
}

// x_f16 rows are x[s][b] at row s*64+b; fill word cols [0,300) and zero pad [556,576)
__global__ void k_gather_words(const int* __restrict__ words, const float* __restrict__ emb,
                               _Float16* __restrict__ x) {
  int i = blockIdx.x * blockDim.x + threadIdx.x;
  const int stride = gridDim.x * blockDim.x;
  const int total = NCH * DIN_PAD;
  for (; i < total; i += stride) {
    const int r = i / DIN_PAD, c = i - r * DIN_PAD;
    if (c < DE)        x[i] = (_Float16)emb[(long)words[r] * DE + c];
    else if (c >= DIN) x[i] = (_Float16)0.0f;   // cols [300,556) written by char LSTM
  }
}

// ch_f16 (W, 4096, 64)
__global__ void k_gather_chars(const int* __restrict__ cidx, const float* __restrict__ emb,
                               _Float16* __restrict__ ch) {
  int i = blockIdx.x * blockDim.x + threadIdx.x;
  const int stride = gridDim.x * blockDim.x;
  const int total = W_LEN * NCH * DC;
  for (; i < total; i += stride) {
    const int r = i >> 6, c = i & (DC - 1);
    ch[i] = (_Float16)emb[cidx[r] * DC + c];
  }
}

// ---------------- char BiLSTM ----------------
// grid (4096/64, 2 dirs) x 256 threads. Per block: 64-row batch tile, full 16-step scan.
// LDS: h ping-pong f16 [2][64][128] (32KB) + c f32 [64][128] (32KB) = 64KB
__global__ void __launch_bounds__(256)
k_char_lstm(const _Float16* __restrict__ ch, const _Float16* __restrict__ cWih_all,
            const _Float16* __restrict__ cWhh_all, const float* __restrict__ cb_f,
            const float* __restrict__ cb_b, _Float16* __restrict__ x) {
  extern __shared__ __align__(16) char smem[];
  _Float16* hbuf = (_Float16*)smem;                         // [2][64][HC]
  float*    cbuf = (float*)(smem + 2 * 64 * HC * sizeof(_Float16));

  const int dir = blockIdx.y;
  const int rowBase = blockIdx.x * 64;
  const int tid = threadIdx.x, wave = tid >> 5, lane = tid & 31;
  const int nl = lane & 15, hi8 = (lane >> 4) << 3;

  const _Float16* Wih = cWih_all + dir * (GC * DC);
  const _Float16* Whh = cWhh_all + dir * (GC * HC);
  const float* bias = dir ? cb_b : cb_f;

  for (int i = tid; i < 64 * HC; i += 256) { cbuf[i] = 0.0f; hbuf[i] = (_Float16)0.0f; }
  __syncthreads();

  for (int t = 0; t < W_LEN; ++t) {
    const int tt = dir ? (W_LEN - 1 - t) : t;
    const int cur = t & 1, nxt = cur ^ 1;
    _Float16* hcur = hbuf + cur * 64 * HC;
    _Float16* hnxt = hbuf + nxt * 64 * HC;

    for (int q = 0; q < 4; ++q) {
      const int task = wave * 4 + q;           // 32 tasks: 4 m-tiles x 8 n-tiles
      const int mt = task >> 3, nt = task & 7;
      v8f acc[4];
      #pragma unroll
      for (int g = 0; g < 4; ++g) {
        const float b = bias[g * HC + nt * 16 + nl];
        #pragma unroll
        for (int r = 0; r < 8; ++r) acc[g][r] = b;
      }
      // input contribution: K = DC = 64
      const _Float16* xt = ch + ((long)tt * NCH + rowBase + mt * 16) * DC;
      #pragma unroll
      for (int kc = 0; kc < DC / 32; ++kc) {
        const v16h a = load_a(xt + kc * 32, DC, lane);
        #pragma unroll
        for (int g = 0; g < 4; ++g) {
          const v16h b = load_b(Wih + (g * HC + nt * 16) * DC + kc * 32, DC, lane);
          acc[g] = __builtin_amdgcn_wmma_f32_16x16x32_f16(false, a, false, b, (short)0, acc[g], false, false);
        }
      }
      // recurrent contribution: K = HC = 128, A from LDS
      #pragma unroll
      for (int kc = 0; kc < HC / 32; ++kc) {
        const _Float16* p = hcur + (mt * 16 + nl) * HC + kc * 32 + hi8;
        v16h a;
        *(float4*)&a       = *(const float4*)(p);
        *((float4*)&a + 1) = *(const float4*)(p + 16);
        #pragma unroll
        for (int g = 0; g < 4; ++g) {
          const v16h b = load_b(Whh + (g * HC + nt * 16) * HC + kc * 32, HC, lane);
          acc[g] = __builtin_amdgcn_wmma_f32_16x16x32_f16(false, a, false, b, (short)0, acc[g], false, false);
        }
      }
      // fused gate nonlinearity + cell update (task-private c patch)
      #pragma unroll
      for (int r = 0; r < 8; ++r) {
        const int m = mt * 16 + r + hi8;
        const int n = nt * 16 + nl;
        const float iv = sigf(acc[0][r]);
        const float fv = sigf(acc[1][r]);
        const float gv = tanhf(acc[2][r]);
        const float ov = sigf(acc[3][r]);
        const float cn = fv * cbuf[m * HC + n] + iv * gv;
        cbuf[m * HC + n] = cn;
        hnxt[m * HC + n] = (_Float16)(ov * tanhf(cn));
      }
    }
    __syncthreads();
  }
  // after 16 steps final h sits in buffer 0; scatter into x_f16 char columns.
  // char row n -> (b = n>>6, s = n&63) -> x row s*64+b  (reshape(B,S).transpose)
  for (int i = tid; i < 64 * HC; i += 256) {
    const int m = i >> 7, c = i & (HC - 1);
    const int n = rowBase + m;
    const int b = n >> 6, s = n & 63;
    x[(s * 64 + b) * DIN_PAD + DE + dir * HC + c] = hbuf[i];
  }
}

// ---------------- word-LSTM input projection ----------------
// P = x_f16 (4096 x 576) @ wWih^T (576 x 2048) + wb -> f32. Shared by both directions.
// 1x4 n-tile register blocking: one A load feeds 4 WMMAs.
__global__ void __launch_bounds__(256)
k_word_proj(const _Float16* __restrict__ x, const _Float16* __restrict__ Wih,
            const float* __restrict__ wb, float* __restrict__ P) {
  const int tid = threadIdx.x, wave = tid >> 5, lane = tid & 31;
  const int nl = lane & 15, hi8 = (lane >> 4) << 3;
  const int id = blockIdx.x * 8 + wave;        // 8192 wave-tasks
  const int mt  = id >> 5;                     // 256 m-tiles
  const int ntg = id & 31;                     // 32 groups of 4 n-tiles
  v8f acc[4];
  #pragma unroll
  for (int j = 0; j < 4; ++j) {
    const float bv = wb[(ntg * 4 + j) * 16 + nl];
    #pragma unroll
    for (int r = 0; r < 8; ++r) acc[j][r] = bv;
  }
  const _Float16* arow = x + (mt * 16) * DIN_PAD;
  for (int kc = 0; kc < DIN_PAD / 32; ++kc) {  // 18 K-chunks
    const v16h a = load_a(arow + kc * 32, DIN_PAD, lane);
    #pragma unroll
    for (int j = 0; j < 4; ++j) {
      const v16h b = load_b(Wih + ((ntg * 4 + j) * 16) * DIN_PAD + kc * 32, DIN_PAD, lane);
      acc[j] = __builtin_amdgcn_wmma_f32_16x16x32_f16(false, a, false, b, (short)0, acc[j], false, false);
    }
  }
  #pragma unroll
  for (int j = 0; j < 4; ++j)
    #pragma unroll
    for (int r = 0; r < 8; ++r)
      P[(mt * 16 + r + hi8) * GW + (ntg * 4 + j) * 16 + nl] = acc[j][r];
}

// ---------------- word BiLSTM scan ----------------
// Recurrence is independent per batch row -> split batch across workgroups.
// grid (4 batch-tiles of 16 rows, 2 dirs) x 256 threads (8 waves). 64 sequential steps.
// LDS: h ping-pong f16 [2][16][512] (32KB) + c f32 [16][512] (32KB) = 64KB
__global__ void __launch_bounds__(256)
k_word_scan(const float* __restrict__ P, const _Float16* __restrict__ Whh,
            float* __restrict__ outH, float* __restrict__ outC) {
  extern __shared__ __align__(16) char smem[];
  _Float16* hbuf = (_Float16*)smem;                         // [2][16][HW]
  float*    cbuf = (float*)(smem + 2 * 16 * HW * sizeof(_Float16));

  const int dir = blockIdx.y;
  const int rowBase = blockIdx.x * 16;         // batch rows handled by this WG
  const int tid = threadIdx.x, wave = tid >> 5, lane = tid & 31;
  const int nl = lane & 15, hi8 = (lane >> 4) << 3;

  for (int i = tid; i < 16 * HW; i += 256) { cbuf[i] = 0.0f; hbuf[i] = (_Float16)0.0f; }
  __syncthreads();

  for (int step = 0; step < S_LEN; ++step) {
    const int s = dir ? (S_LEN - 1 - step) : step;
    const int cur = step & 1, nxt = cur ^ 1;
    _Float16* hcur = hbuf + cur * 16 * HW;
    _Float16* hnxt = hbuf + nxt * 16 * HW;
    const float* Pt = P + ((long)s * 64 + rowBase) * GW;   // 16 rows of gates

    for (int q = 0; q < 4; ++q) {
      const int nt = wave * 4 + q;             // 32 n-tiles, single m-tile
      v8f acc[4];
      #pragma unroll
      for (int g = 0; g < 4; ++g)
        #pragma unroll
        for (int r = 0; r < 8; ++r)
          acc[g][r] = Pt[(r + hi8) * GW + g * HW + nt * 16 + nl];

      for (int kc = 0; kc < HW / 32; ++kc) {   // 16 K-chunks over h
        __builtin_prefetch(Whh + (nt * 16 + nl) * HW + ((kc + 1) & 15) * 32, 0, 0);
        const _Float16* p = hcur + nl * HW + kc * 32 + hi8;
        v16h a;
        *(float4*)&a       = *(const float4*)(p);
        *((float4*)&a + 1) = *(const float4*)(p + 16);
        #pragma unroll
        for (int g = 0; g < 4; ++g) {
          const v16h b = load_b(Whh + (g * HW + nt * 16) * HW + kc * 32, HW, lane);
          acc[g] = __builtin_amdgcn_wmma_f32_16x16x32_f16(false, a, false, b, (short)0, acc[g], false, false);
        }
      }
      #pragma unroll
      for (int r = 0; r < 8; ++r) {
        const int m = r + hi8;                 // local batch row
        const int n = nt * 16 + nl;            // hidden unit
        const float iv = sigf(acc[0][r]);
        const float fv = sigf(acc[1][r]);
        const float gv = tanhf(acc[2][r]);
        const float ov = sigf(acc[3][r]);
        const float cn = fv * cbuf[m * HW + n] + iv * gv;
        cbuf[m * HW + n] = cn;
        const float hn = ov * tanhf(cn);
        hnxt[m * HW + n] = (_Float16)hn;
        const long orow = (long)(rowBase + m) * S_LEN + s;  // output row = b*S + s
        outH[orow * (2 * HW) + dir * HW + n] = hn;
        outC[orow * (2 * HW) + dir * HW + n] = cn;
      }
    }
    __syncthreads();
  }
}

extern "C" void kernel_launch(void* const* d_in, const int* in_sizes, int n_in,
                              void* d_out, int out_size, void* d_ws, size_t ws_size,
                              hipStream_t stream) {
  (void)in_sizes; (void)n_in; (void)out_size; (void)ws_size;
  const int*   words  = (const int*)d_in[0];
  const int*   chars  = (const int*)d_in[1];
  const float* wemb   = (const float*)d_in[2];
  const float* cemb   = (const float*)d_in[3];
  const float* cWih_f = (const float*)d_in[4];
  const float* cWhh_f = (const float*)d_in[5];
  const float* cbf    = (const float*)d_in[6];
  const float* cWih_b = (const float*)d_in[7];
  const float* cWhh_b = (const float*)d_in[8];
  const float* cbb    = (const float*)d_in[9];
  const float* wWih   = (const float*)d_in[10];
  const float* wWhh   = (const float*)d_in[11];
  const float* wb     = (const float*)d_in[12];

  char* ws = (char*)d_ws;
  size_t off = 0;
  auto alloc = [&](size_t bytes) { char* p = ws + off; off += (bytes + 255) & ~(size_t)255; return p; };
  _Float16* wWih16 = (_Float16*)alloc((size_t)GW * DIN_PAD * 2);
  _Float16* wWhh16 = (_Float16*)alloc((size_t)GW * HW * 2);
  _Float16* cWih16 = (_Float16*)alloc((size_t)2 * GC * DC * 2);
  _Float16* cWhh16 = (_Float16*)alloc((size_t)2 * GC * HC * 2);
  _Float16* x16    = (_Float16*)alloc((size_t)NCH * DIN_PAD * 2);
  _Float16* ch16   = (_Float16*)alloc((size_t)W_LEN * NCH * DC * 2);
  float*    Pbuf   = (float*)   alloc((size_t)NCH * GW * 4);

  // weight conversion / packing
  k_cvt<<<512, 256, 0, stream>>>(wWhh,   wWhh16,            GW * HW);
  k_cvt<<<64,  256, 0, stream>>>(cWih_f, cWih16,            GC * DC);
  k_cvt<<<64,  256, 0, stream>>>(cWih_b, cWih16 + GC * DC,  GC * DC);
  k_cvt<<<128, 256, 0, stream>>>(cWhh_f, cWhh16,            GC * HC);
  k_cvt<<<128, 256, 0, stream>>>(cWhh_b, cWhh16 + GC * HC,  GC * HC);
  k_pack_wWih<<<1024, 256, 0, stream>>>(wWih, wWih16);

  // embedding gathers
  k_gather_words<<<2048, 256, 0, stream>>>(words, wemb, x16);
  k_gather_chars<<<2048, 256, 0, stream>>>(chars, cemb, ch16);

  // char BiLSTM: writes char features into x16[:, 300:556)
  dim3 gc(NCH / 64, 2);
  k_char_lstm<<<gc, 256, 64 * 1024, stream>>>(ch16, cWih16, cWhh16, cbf, cbb, x16);

  // shared input projection for both word-LSTM directions (1x4 n-blocked)
  k_word_proj<<<1024, 256, 0, stream>>>(x16, wWih16, wb, Pbuf);

  // sequential bidirectional word scan, batch-parallel across WGs
  float* outH = (float*)d_out;
  float* outC = outH + (size_t)NCH * 2 * HW;
  dim3 gs(4, 2);
  k_word_scan<<<gs, 256, 64 * 1024, stream>>>(Pbuf, wWhh16, outH, outC);
}
